// Predictor6_17274358465190
// MI455X (gfx1250) — compile-verified, matrix operand order checked by reading, not run on hardware
//
#include <hip/hip_runtime.h>
#include <math.h>

#define N_NODES  20000
#define N_EDGES  320000
#define E_TOT    (N_EDGES + N_NODES)
#define N_GRAPHS 128
#define IN_CH    64
#define DIM      32
#define NEG_SLOPE 0.2f
#define EPS_F     1e-16f

typedef __attribute__((ext_vector_type(16))) __bf16 v16bf;
typedef __attribute__((ext_vector_type(8)))  float  v8f;
typedef __attribute__((ext_vector_type(4)))  unsigned int u32x4;
typedef __attribute__((ext_vector_type(8)))  int i32x8;
typedef __attribute__((ext_vector_type(4)))  int i32x4;

#if __has_builtin(__builtin_amdgcn_tensor_load_to_lds)
#define USE_TDM 1
#endif

union FragAB { v16bf v; unsigned int u[8]; };

// pack two f32 -> packed bf16x2 (low = a, high = b)
__device__ __forceinline__ unsigned int pack2_bf16(float a, float b) {
#if __has_builtin(__builtin_amdgcn_cvt_pk_bf16_f32)
    typedef __attribute__((ext_vector_type(2))) __bf16 bf16x2;
    union { bf16x2 v; unsigned int u; } cv;
    cv.v = __builtin_amdgcn_cvt_pk_bf16_f32(a, b);
    return cv.u;
#else
    // native fptrunc: backend selects packed cvt when available (RNE either way)
    union { __bf16 h[2]; unsigned int u; } cv;
    cv.h[0] = (__bf16)a;
    cv.h[1] = (__bf16)b;
    return cv.u;
#endif
}

// Orderable encoding of float for u32 atomicMax
__device__ __forceinline__ unsigned int enc_f(float f) {
    unsigned int u = __float_as_uint(f);
    return (u >> 31) ? ~u : (u | 0x80000000u);
}
__device__ __forceinline__ float dec_f(unsigned int u) {
    return __uint_as_float((u >> 31) ? (u & 0x7FFFFFFFu) : ~u);
}

__device__ __forceinline__ void edge_sd(const int* __restrict__ ei, int e, int& s, int& d) {
    if (e < N_EDGES) { s = ei[e]; d = ei[N_EDGES + e]; }
    else             { s = e - N_EDGES; d = s; }        // appended self-loops
}

// Low 32 bits of a generic pointer to LDS == LDS byte offset (aperture mapping)
__device__ __forceinline__ unsigned lds_off32(const void* p) {
    return (unsigned)(unsigned long long)(size_t)p;
}

#ifdef USE_TDM
// Issue one TDM 2D tile load: global (row-major, f32) -> LDS.
// tensor_d0/d1 = remaining tensor extent (elements) for OOB zero-fill,
// stride0 = row stride (elements), tile = tile_d0 x tile_d1 elements.
__device__ __forceinline__ void tdm_load_2d(
    const float* gaddr, unsigned lds_off,
    unsigned tensor_d0, unsigned tensor_d1,
    unsigned stride0, unsigned tile_d0, unsigned tile_d1)
{
    unsigned long long ga = (unsigned long long)(size_t)gaddr;
    u32x4 g0;
    g0[0] = 1u;                                         // count=1, user mode
    g0[1] = lds_off;                                    // lds_addr
    g0[2] = (unsigned)(ga & 0xFFFFFFFFu);               // global_addr[31:0]
    g0[3] = (unsigned)((ga >> 32) & 0x01FFFFFFu)        // global_addr[56:32]
          | (2u << 30);                                 // type=2 ("image")
    i32x8 g1;
    g1[0] = (int)(2u << 16);                            // wg_mask=0, data_size=4B
    g1[1] = (int)((tensor_d0 & 0xFFFFu) << 16);         // tensor_dim0[15:0]
    g1[2] = (int)((tensor_d0 >> 16) | ((tensor_d1 & 0xFFFFu) << 16));
    g1[3] = (int)((tensor_d1 >> 16) | (tile_d0 << 16)); // tile_dim0
    g1[4] = (int)(tile_d1);                             // tile_dim1 (tile_dim2=0)
    g1[5] = (int)(stride0);                             // tensor_dim0_stride[31:0]
    g1[6] = 0;
    g1[7] = 0;
    i32x4 z4 = {0, 0, 0, 0};
#if __clang_major__ >= 23
    i32x8 z8 = {0, 0, 0, 0, 0, 0, 0, 0};
    __builtin_amdgcn_tensor_load_to_lds(g0, g1, z4, z4, z8, 0);
#else
    __builtin_amdgcn_tensor_load_to_lds(g0, g1, z4, z4, 0);
#endif
}

__device__ __forceinline__ void tdm_wait0() {
#if __has_builtin(__builtin_amdgcn_s_wait_tensorcnt)
    __builtin_amdgcn_s_wait_tensorcnt(0);
#else
    asm volatile("s_wait_tensorcnt 0x0" ::: "memory");
#endif
}
#endif // USE_TDM

// ---------------------------------------------------------------------------
// Tiled bf16 WMMA GEMM:  H[N,M] = X[N,K] * W[M,K]^T   (f32 in/out, bf16 MACs)
// Block: 256 threads (8 waves). Tile: 128 rows x 32 cols, K-step 32.
// TDM stages f32 tiles into LDS; cooperative one-pass f32->bf16 convert in
// LDS; fragments assembled from bf16 LDS; v_wmma_f32_16x16x32_bf16 MACs.
// Requires K % 32 == 0 and M % 32 == 0 (true for all call sites).
// ---------------------------------------------------------------------------
__global__ __launch_bounds__(256) void gemm_xwt_bf16(
    const float* __restrict__ X, const float* __restrict__ W,
    float* __restrict__ Hd, int N, int K, int M)
{
    __shared__ float          ldsAf[128][32];   // 16 KB f32 staging (TDM dest)
    __shared__ float          ldsBf[32][32];    //  4 KB
    __shared__ unsigned short ldsA16[128][34];  // 8.5 KB bf16 (+pad)
    __shared__ unsigned short ldsB16[32][34];   // 2.1 KB

    const int t    = threadIdx.x;
    const int lane = t & 31;
    const int wave = t >> 5;
    const int half = lane >> 4;
    const int l16  = lane & 15;
    const int rowBase = blockIdx.x * 128;
    const int colBase = blockIdx.y * 32;

    v8f acc0 = {};
    v8f acc1 = {};

    for (int kk = 0; kk < K; kk += 32) {
#ifdef USE_TDM
        if (wave == 0) {
            // A tile: 128 rows x 32 k; OOB rows (>= N-rowBase) zero-filled by TDM
            tdm_load_2d(X + (size_t)rowBase * K + kk, lds_off32(&ldsAf[0][0]),
                        (unsigned)(K - kk), (unsigned)(N - rowBase),
                        (unsigned)K, 32u, 128u);
            // B tile: 32 output-cols x 32 k from W rows (M multiple of 32)
            tdm_load_2d(W + (size_t)colBase * K + kk, lds_off32(&ldsBf[0][0]),
                        (unsigned)(K - kk), (unsigned)(M - colBase),
                        (unsigned)K, 32u, 32u);
            tdm_wait0();
        }
#else
        // Fallback: synchronous float4 staging
        #pragma unroll
        for (int i = 0; i < 4; ++i) {
            int idx = t + i * 256;               // 1024 float4 for A
            int r   = idx >> 3;
            int k0  = (idx & 7) << 2;
            int gr  = rowBase + r;
            float4 f = make_float4(0.f, 0.f, 0.f, 0.f);
            if (gr < N) f = *(const float4*)(X + (size_t)gr * K + kk + k0);
            *(float4*)&ldsAf[r][k0] = f;
        }
        {
            int r  = t >> 3;
            int k0 = (t & 7) << 2;
            float4 f = *(const float4*)(W + (size_t)(colBase + r) * K + kk + k0);
            *(float4*)&ldsBf[r][k0] = f;
        }
#endif
        __syncthreads();   // f32 tiles visible

        // Cooperative f32 -> bf16 convert (each element converted exactly once)
        #pragma unroll
        for (int i = 0; i < 8; ++i) {           // A: 2048 bf16x2 pairs
            int p  = t + i * 256;
            int r  = p >> 4;
            int kp = (p & 15) << 1;
            float2 f = *(const float2*)&ldsAf[r][kp];
            *(unsigned int*)&ldsA16[r][kp] = pack2_bf16(f.x, f.y);
        }
        #pragma unroll
        for (int i = 0; i < 2; ++i) {           // B: 512 bf16x2 pairs
            int p  = t + i * 256;
            int r  = p >> 4;
            int kp = (p & 15) << 1;
            float2 f = *(const float2*)&ldsBf[r][kp];
            *(unsigned int*)&ldsB16[r][kp] = pack2_bf16(f.x, f.y);
        }
        __syncthreads();   // bf16 tiles visible

        // Fragments per ISA 16-bit 16x32 layout (K pairs contiguous -> b32 reads)
        FragAB a, b0, b1;
        const int mr = (wave << 4) + l16;
        #pragma unroll
        for (int j = 0; j < 8; ++j) {
            int kb = (j < 4) ? (half * 8 + 2 * j) : (16 + half * 8 + 2 * (j - 4));
            a.u[j]  = *(const unsigned int*)&ldsA16[mr][kb];
            b0.u[j] = *(const unsigned int*)&ldsB16[l16][kb];
            b1.u[j] = *(const unsigned int*)&ldsB16[16 + l16][kb];
        }
        acc0 = __builtin_amdgcn_wmma_f32_16x16x32_bf16(false, a.v, false, b0.v,
                                                       (short)0, acc0, false, false);
        acc1 = __builtin_amdgcn_wmma_f32_16x16x32_bf16(false, a.v, false, b1.v,
                                                       (short)0, acc1, false, false);
        // no trailing barrier needed: wmma phase reads only bf16 tiles, which
        // are next written after the post-TDM barrier of the next iteration
    }

    // C/D layout: element r at lane l -> row = r + 8*(l>>4), col = l&15
    const int rbase = rowBase + (wave << 4) + half * 8;
    #pragma unroll
    for (int r = 0; r < 8; ++r) {
        int row = rbase + r;
        if (row < N) {
            Hd[(size_t)row * M + colBase + l16]      = acc0[r];
            Hd[(size_t)row * M + colBase + 16 + l16] = acc1[r];
        }
    }
}

__global__ __launch_bounds__(256) void attn_proj(
    const float* __restrict__ h, const float* __restrict__ atts,
    const float* __restrict__ attd, float* __restrict__ asrc,
    float* __restrict__ adst, int Hh, int C)
{
    int widx = (blockIdx.x << 3) + (threadIdx.x >> 5);
    int lane = threadIdx.x & 31;
    if (widx >= N_NODES * Hh) return;
    int n = widx / Hh, hd = widx - n * Hh;
    const float* hp = h + (size_t)n * Hh * C + hd * C;
    const float* sp = atts + hd * C;
    const float* dp = attd + hd * C;
    float ss = 0.f, dd = 0.f;
    for (int c = lane; c < C; c += 32) {
        float v = hp[c];
        ss += v * sp[c];
        dd += v * dp[c];
    }
    #pragma unroll
    for (int o = 16; o > 0; o >>= 1) {
        ss += __shfl_xor(ss, o, 32);
        dd += __shfl_xor(dd, o, 32);
    }
    if (lane == 0) { asrc[widx] = ss; adst[widx] = dd; }
}

__global__ void fill_u32(unsigned int* __restrict__ p, unsigned int v, int n) {
    int i = blockIdx.x * blockDim.x + threadIdx.x;
    if (i < n) p[i] = v;
}

__global__ void init_bias(float* __restrict__ out, const float* __restrict__ b,
                          int HC, int total) {
    int i = blockIdx.x * blockDim.x + threadIdx.x;
    if (i < total) out[i] = b[i % HC];
}

__global__ void edge_score_max(const int* __restrict__ ei, const float* __restrict__ asrc,
                               const float* __restrict__ adst, float* __restrict__ ebuf,
                               unsigned int* __restrict__ mu, int Hh)
{
    int tid = blockIdx.x * blockDim.x + threadIdx.x;
    if (tid >= E_TOT * Hh) return;
    int e = tid / Hh, hd = tid - e * Hh;
    int s, d; edge_sd(ei, e, s, d);
    float v = asrc[s * Hh + hd] + adst[d * Hh + hd];
    v = (v > 0.f) ? v : NEG_SLOPE * v;                 // leaky_relu
    ebuf[tid] = v;
    atomicMax(&mu[d * Hh + hd], enc_f(v));             // segment max
}

__global__ void edge_exp_sum(const int* __restrict__ ei, float* __restrict__ ebuf,
                             const unsigned int* __restrict__ mu,
                             float* __restrict__ ssum, int Hh)
{
    int tid = blockIdx.x * blockDim.x + threadIdx.x;
    if (tid >= E_TOT * Hh) return;
    int e = tid / Hh, hd = tid - e * Hh;
    int s, d; edge_sd(ei, e, s, d);
    float m  = dec_f(mu[d * Hh + hd]);
    float ex = __expf(ebuf[tid] - m);
    ebuf[tid] = ex;
    atomicAdd(&ssum[d * Hh + hd], ex);                 // segment sum
}

__global__ void edge_aggregate(const int* __restrict__ ei, const float* __restrict__ ebuf,
                               const float* __restrict__ ssum, const float* __restrict__ h,
                               float* __restrict__ out, int Hh, int C)
{
    long long tid = (long long)blockIdx.x * blockDim.x + threadIdx.x;
    long long total = (long long)E_TOT * Hh * C;
    if (tid >= total) return;
    int c    = (int)(tid % C);
    int rest = (int)(tid / C);                          // = e*Hh + hd
    int hd   = rest % Hh;
    int e    = rest / Hh;
    int s, d; edge_sd(ei, e, s, d);
    float alpha = ebuf[rest] / (ssum[d * Hh + hd] + EPS_F);
    atomicAdd(&out[(size_t)d * Hh * C + hd * C + c],
              alpha * h[(size_t)s * Hh * C + hd * C + c]);
}

__global__ void relu_k(float* __restrict__ p, int n) {
    int i = blockIdx.x * blockDim.x + threadIdx.x;
    if (i < n) p[i] = fmaxf(p[i], 0.f);
}

__global__ void pool_k(const float* __restrict__ h, const int* __restrict__ batch,
                       float* __restrict__ g)
{
    int i = blockIdx.x * blockDim.x + threadIdx.x;
    if (i >= N_NODES * DIM) return;
    int n = i >> 5, c = i & 31;
    atomicAdd(&g[batch[n] * DIM + c], h[i]);
}

__global__ void bias_relu(float* __restrict__ p, const float* __restrict__ b,
                          int M, int total) {
    int i = blockIdx.x * blockDim.x + threadIdx.x;
    if (i < total) p[i] = fmaxf(p[i] + b[i % M], 0.f);
}

__global__ __launch_bounds__(256) void lin2_sigmoid(
    const float* __restrict__ x, const float* __restrict__ w,
    const float* __restrict__ b, float* __restrict__ out)
{
    __shared__ float red[8];
    const float* xp = x + blockIdx.x * 1024;
    float acc = 0.f;
    for (int i = threadIdx.x; i < 1024; i += 256) acc += xp[i] * w[i];
    #pragma unroll
    for (int o = 16; o > 0; o >>= 1) acc += __shfl_xor(acc, o, 32);
    if ((threadIdx.x & 31) == 0) red[threadIdx.x >> 5] = acc;
    __syncthreads();
    if (threadIdx.x == 0) {
        float s = 0.f;
        #pragma unroll
        for (int i = 0; i < 8; ++i) s += red[i];
        out[blockIdx.x] = 1.f / (1.f + __expf(-(s + b[0])));
    }
}

static void run_gat_layer(hipStream_t stream, const float* xin, int K,
                          const float* W, const float* as_, const float* ad_,
                          const float* b_, int Hh, int C,
                          float* hlin, float* outp,
                          float* asrc, float* adst, unsigned int* mu,
                          float* ssum, float* ebuf, const int* ei)
{
    const int HC = Hh * C;
    dim3 gg((N_NODES + 127) / 128, HC / 32);
    gemm_xwt_bf16<<<gg, 256, 0, stream>>>(xin, W, hlin, N_NODES, K, HC);

    int nw = N_NODES * Hh;
    attn_proj<<<(nw + 7) / 8, 256, 0, stream>>>(hlin, as_, ad_, asrc, adst, Hh, C);
    fill_u32<<<(nw + 255) / 256, 256, 0, stream>>>(mu, 0x007FFFFFu, nw);   // enc(-inf)
    fill_u32<<<(nw + 255) / 256, 256, 0, stream>>>((unsigned int*)ssum, 0u, nw);

    int tot = N_NODES * HC;
    init_bias<<<(tot + 255) / 256, 256, 0, stream>>>(outp, b_, HC, tot);

    int et = E_TOT * Hh;
    edge_score_max<<<(et + 255) / 256, 256, 0, stream>>>(ei, asrc, adst, ebuf, mu, Hh);
    edge_exp_sum<<<(et + 255) / 256, 256, 0, stream>>>(ei, ebuf, mu, ssum, Hh);

    long long etot = (long long)E_TOT * HC;
    int eb = (int)((etot + 255) / 256);
    edge_aggregate<<<eb, 256, 0, stream>>>(ei, ebuf, ssum, hlin, outp, Hh, C);

    relu_k<<<(tot + 255) / 256, 256, 0, stream>>>(outp, tot);
}

extern "C" void kernel_launch(void* const* d_in, const int* in_sizes, int n_in,
                              void* d_out, int out_size, void* d_ws, size_t ws_size,
                              hipStream_t stream)
{
    const float* x     = (const float*)d_in[0];
    const float* W1    = (const float*)d_in[1];
    const float* as1   = (const float*)d_in[2];
    const float* ad1   = (const float*)d_in[3];
    const float* b1    = (const float*)d_in[4];
    const float* W2    = (const float*)d_in[5];
    const float* as2   = (const float*)d_in[6];
    const float* ad2   = (const float*)d_in[7];
    const float* b2    = (const float*)d_in[8];
    const float* W3    = (const float*)d_in[9];
    const float* as3   = (const float*)d_in[10];
    const float* ad3   = (const float*)d_in[11];
    const float* b3    = (const float*)d_in[12];
    const float* lin_w = (const float*)d_in[13];
    const float* lin_b = (const float*)d_in[14];
    const float* l2w   = (const float*)d_in[15];
    const float* l2b   = (const float*)d_in[16];
    const int*   ei    = (const int*)d_in[17];
    const int*   batch = (const int*)d_in[18];

    char* ws = (char*)d_ws;
    size_t off = 0;
    auto alloc = [&](size_t bytes) -> void* {
        void* p = ws + off;
        off = (off + bytes + 255) & ~(size_t)255;
        return p;
    };
    float* hlin1 = (float*)alloc((size_t)N_NODES * 320 * 4);
    float* out1  = (float*)alloc((size_t)N_NODES * 320 * 4);
    float* hlin2 = (float*)alloc((size_t)N_NODES * 480 * 4);
    float* out2  = (float*)alloc((size_t)N_NODES * 480 * 4);
    float* hlin3 = (float*)alloc((size_t)N_NODES * 32 * 4);
    float* out3  = (float*)alloc((size_t)N_NODES * 32 * 4);
    float* asrc  = (float*)alloc((size_t)N_NODES * 5 * 4);
    float* adst  = (float*)alloc((size_t)N_NODES * 5 * 4);
    unsigned int* mu = (unsigned int*)alloc((size_t)N_NODES * 5 * 4);
    float* ssum  = (float*)alloc((size_t)N_NODES * 5 * 4);
    float* ebuf  = (float*)alloc((size_t)E_TOT * 5 * 4);
    float* gpool = (float*)alloc((size_t)N_GRAPHS * DIM * 4);
    float* lin1o = (float*)alloc((size_t)N_GRAPHS * 1024 * 4);

    run_gat_layer(stream, x,    IN_CH, W1, as1, ad1, b1, 5, 64, hlin1, out1,
                  asrc, adst, mu, ssum, ebuf, ei);
    run_gat_layer(stream, out1, 320,   W2, as2, ad2, b2, 5, 96, hlin2, out2,
                  asrc, adst, mu, ssum, ebuf, ei);
    run_gat_layer(stream, out2, 480,   W3, as3, ad3, b3, 1, 32, hlin3, out3,
                  asrc, adst, mu, ssum, ebuf, ei);

    // global_add_pool
    int gn = N_GRAPHS * DIM;
    fill_u32<<<(gn + 255) / 256, 256, 0, stream>>>((unsigned int*)gpool, 0u, gn);
    pool_k<<<(N_NODES * DIM + 255) / 256, 256, 0, stream>>>(out3, batch, gpool);

    // lin1 (WMMA GEMM 128x32x1024) + bias + ReLU
    dim3 gl(1, 1024 / 32);
    gemm_xwt_bf16<<<gl, 256, 0, stream>>>(gpool, lin_w, lin1o, N_GRAPHS, DIM, 1024);
    bias_relu<<<(N_GRAPHS * 1024 + 255) / 256, 256, 0, stream>>>(lin1o, lin_b, 1024,
                                                                 N_GRAPHS * 1024);
    // lin2 + sigmoid -> d_out [128,1]
    lin2_sigmoid<<<N_GRAPHS, 256, 0, stream>>>(lin1o, l2w, l2b, (float*)d_out);
}
